// Playlist_Constructor_70729521430757
// MI455X (gfx1250) — matrix operand order repaired and checked by reading, not compiled, not run on hardware
//
#include <hip/hip_runtime.h>

// ---------------------------------------------------------------------------
// Problem: B=32, S=2048, D=256
// out[b,:] = (1/length[b]) * sum_t ( sum_s softmax_row_s(QK^T)[t] ) * v[t,:]
// ---------------------------------------------------------------------------
#define BATCH 32
#define SEQ   2048
#define DIM   256
#define NEGV  (-1e9f)
#define QBLK  128                 // queries per workgroup (8 waves x 16)
#define KBLK  64                  // keys per LDS tile
#define NQB   (SEQ / QBLK)        // 16 query blocks per batch
#define NKB   (SEQ / KBLK)        // 32 key blocks

typedef __attribute__((ext_vector_type(16))) __bf16          v16bf;
typedef __attribute__((ext_vector_type(8)))  float           v8f;
typedef __attribute__((ext_vector_type(8)))  unsigned short  us8;
typedef __attribute__((ext_vector_type(16))) unsigned short  us16;
typedef __attribute__((ext_vector_type(4)))  unsigned int    ui4;
typedef __attribute__((ext_vector_type(8)))  int             i8v;
typedef __attribute__((ext_vector_type(4)))  int             i4v;
typedef unsigned short u16;

#if defined(__has_builtin)
#if __has_builtin(__builtin_amdgcn_tensor_load_to_lds)
#define HAVE_TDM 1
#endif
#endif

// native fp32 -> bf16 (v_cvt), bf16 -> fp32
__device__ __forceinline__ u16 f2bf(float f) {
  __bf16 h = (__bf16)f;
  return __builtin_bit_cast(u16, h);
}
__device__ __forceinline__ float bf2f(u16 h) {
  unsigned u = ((unsigned)h) << 16;
  return __builtin_bit_cast(float, u);
}
__device__ __forceinline__ v16bf combine8(us8 lo, us8 hi) {
  us16 u = __builtin_shufflevector(lo, hi, 0,1,2,3,4,5,6,7,8,9,10,11,12,13,14,15);
  return __builtin_bit_cast(v16bf, u);
}

// ---------------------------------------------------------------------------
// Kernel 0: bulk fp32 -> bf16 conversion (8 elements / thread)
// ---------------------------------------------------------------------------
__global__ void f32_to_bf16_kernel(const float* __restrict__ in,
                                   u16* __restrict__ out) {
  size_t i = ((size_t)blockIdx.x * 256 + threadIdx.x) * 8;
  float4 a = *(const float4*)(in + i);
  float4 b = *(const float4*)(in + i + 4);
  us8 o;
  o[0]=f2bf(a.x); o[1]=f2bf(a.y); o[2]=f2bf(a.z); o[3]=f2bf(a.w);
  o[4]=f2bf(b.x); o[5]=f2bf(b.y); o[6]=f2bf(b.z); o[7]=f2bf(b.w);
  *(us8*)(out + i) = o;
}

// ---------------------------------------------------------------------------
// Tensor Data Mover: async copy of one contiguous 64x256 bf16 K-tile (32 KB)
// into LDS. Descriptor built per CDNA5 ISA D# layout (1-D tile).
// ---------------------------------------------------------------------------
#ifdef HAVE_TDM
__device__ __forceinline__ void tdm_load_ktile(u16* lds_dst, const u16* gsrc) {
  const unsigned elems = KBLK * DIM;                       // 16384 bf16
  unsigned long long ga = (unsigned long long)(uintptr_t)gsrc;
  unsigned lds = (unsigned)(uintptr_t)lds_dst;             // LDS offset = low 32b
  ui4 g0;
  g0[0] = 1u;                                              // count=1 (valid user D#)
  g0[1] = lds;                                             // lds_addr
  g0[2] = (unsigned)ga;                                    // global_addr[31:0]
  g0[3] = (unsigned)((ga >> 32) & 0x01FFFFFFu) | 0x80000000u; // addr[56:32] | type=2
  i8v g1;
  g1[0] = 0x00010000;                                      // data_size=1 (2 bytes)
  g1[1] = (int)((elems & 0xFFFFu) << 16);                  // tensor_dim0[15:0]
  g1[2] = (int)(((elems >> 16) & 0xFFFFu) | (1u << 16));   // tensor_dim0[31:16], tensor_dim1=1
  g1[3] = (int)((elems & 0xFFFFu) << 16);                  // tile_dim0 = elems
  g1[4] = 0;                                               // tile_dim1/2 unused
  g1[5] = (int)elems;                                      // tensor_dim0_stride
  g1[6] = 0;
  g1[7] = 0;
  i4v z4 = {0, 0, 0, 0};
#if __clang_major__ >= 23
  i8v z8 = {0, 0, 0, 0, 0, 0, 0, 0};
  __builtin_amdgcn_tensor_load_to_lds(g0, g1, z4, z4, z8, 0);
#else
  __builtin_amdgcn_tensor_load_to_lds(g0, g1, z4, z4, 0);
#endif
}
#endif

__device__ __forceinline__ void load_ktile(u16* dst, const u16* src,
                                           int tid, int wave) {
#ifdef HAVE_TDM
  if (wave == 0) tdm_load_ktile(dst, src);
#else
  const uint4* s = (const uint4*)src;
  uint4* d = (uint4*)dst;
  for (int i = tid; i < (KBLK * DIM) / 8; i += 256) d[i] = s[i];
#endif
}
__device__ __forceinline__ void wait_ktile_pend1(int wave) {
#ifdef HAVE_TDM
  if (wave == 0) __builtin_amdgcn_s_wait_tensorcnt(1);
#endif
}
__device__ __forceinline__ void wait_ktile_pend0(int wave) {
#ifdef HAVE_TDM
  if (wave == 0) __builtin_amdgcn_s_wait_tensorcnt(0);
#endif
}

// ---------------------------------------------------------------------------
// Kernel 1: QKV projection (bf16 in, bf16 out, f32 WMMA accumulate)
// grid: (B*3, SEQ/64, DIM/64), block 128 (4 waves). Wave tile: 16 rows x 64 cols.
// ---------------------------------------------------------------------------
__global__ void qkv_proj_kernel(const u16* __restrict__ xb16,
                                const u16* __restrict__ wq16,
                                const u16* __restrict__ wk16,
                                const u16* __restrict__ wv16,
                                u16* __restrict__ qo,
                                u16* __restrict__ ko,
                                u16* __restrict__ vo) {
  const int lane = threadIdx.x & 31;
  const int wave = threadIdx.x >> 5;
  const int b    = blockIdx.x / 3;
  const int mat  = blockIdx.x % 3;
  const u16* W   = (mat == 0) ? wq16 : (mat == 1) ? wk16 : wv16;
  u16*       out = (mat == 0) ? qo   : (mat == 1) ? ko   : vo;

  const int s0 = blockIdx.y * 64 + wave * 16;
  const int e0 = blockIdx.z * 64;
  const int m  = lane & 15;
  const int h  = lane >> 4;

  const u16* xb = xb16 + (size_t)b * SEQ * DIM;
  u16*       ob = out  + (size_t)b * SEQ * DIM;

  v8f acc[4];
  for (int j = 0; j < 4; ++j)
    for (int r = 0; r < 8; ++r) acc[j][r] = 0.0f;

  for (int kd = 0; kd < DIM; kd += 32) {
    const u16* xrow = xb + (size_t)(s0 + m) * DIM + kd;
    v16bf afrag = combine8(*(const us8*)(xrow + h * 16),
                           *(const us8*)(xrow + h * 16 + 8));
    for (int j = 0; j < 4; ++j) {
      const u16* wrow = W + (size_t)(e0 + 16 * j + m) * DIM + kd + h * 16;
      v16bf bfrag = combine8(*(const us8*)(wrow), *(const us8*)(wrow + 8));
      acc[j] = __builtin_amdgcn_wmma_f32_16x16x32_bf16(
          false, afrag, false, bfrag, (short)0, acc[j], false, false);
    }
  }
  for (int j = 0; j < 4; ++j)
    for (int r = 0; r < 8; ++r)
      ob[(size_t)(s0 + r + 8 * h) * DIM + e0 + 16 * j + m] = f2bf(acc[j][r]);
}

// ---------------------------------------------------------------------------
// Kernel 2: flash-style softmax column sums with TDM double-buffered K tiles.
// grid: (B, NQB), block 256 (8 waves x 16 queries each).
// ---------------------------------------------------------------------------
__global__ void flash_colsum_kernel(const u16* __restrict__ qb16,
                                    const u16* __restrict__ kb16,
                                    const unsigned char* __restrict__ mask,
                                    float* __restrict__ partial) {
  __shared__ __align__(16) u16  ktile[2][KBLK * DIM];   // 2 x 32 KB
  __shared__ float colsum[8][KBLK];                     // 2 KB

  const int tid  = threadIdx.x;
  const int lane = tid & 31;
  const int wave = tid >> 5;
  const int b    = blockIdx.x;
  const int qb   = blockIdx.y;
  const int s0   = qb * QBLK + wave * 16;
  const int m    = lane & 15;
  const int h    = lane >> 4;
  const float scale = 0.0625f;   // D^-0.5

  const u16* qbase = qb16 + (size_t)b * SEQ * DIM;
  const u16* kbase = kb16 + (size_t)b * SEQ * DIM;
  const unsigned char* mb = mask + (size_t)b * SEQ;

  // 16 query rows as 8 A-fragments (full D=256)
  v16bf qfrag[8];
  for (int kc = 0; kc < 8; ++kc) {
    const u16* qr = qbase + (size_t)(s0 + m) * DIM + kc * 32;
    qfrag[kc] = combine8(*(const us8*)(qr + h * 16),
                         *(const us8*)(qr + h * 16 + 8));
  }

  float mrow[8], lrow[8];
  for (int r = 0; r < 8; ++r) { mrow[r] = -3.0e38f; lrow[r] = 0.0f; }

  // ----- Pass A: per-row max & softmax denominator -----
  load_ktile(ktile[0], kbase, tid, wave);
  for (int i = 0; i < NKB; ++i) {
    const int t0 = i * KBLK;
    if (i + 1 < NKB) {
      load_ktile(ktile[(i + 1) & 1], kbase + (size_t)(t0 + KBLK) * DIM, tid, wave);
      wait_ktile_pend1(wave);
    } else {
      wait_ktile_pend0(wave);
    }
    __syncthreads();
    const u16* kt_base = ktile[i & 1];

    for (int nt = 0; nt < 4; ++nt) {
      v8f c;
      for (int r = 0; r < 8; ++r) c[r] = 0.0f;
      const u16* kt = kt_base + (nt * 16 + m) * DIM;
      for (int kc = 0; kc < 8; ++kc) {
        const u16* kr = kt + kc * 32 + h * 16;
        v16bf bfrag = combine8(*(const us8*)(kr), *(const us8*)(kr + 8));
        c = __builtin_amdgcn_wmma_f32_16x16x32_bf16(
            false, qfrag[kc], false, bfrag, (short)0, c, false, false);
      }
      const bool msk = mb[t0 + nt * 16 + m] != 0;
      for (int r = 0; r < 8; ++r) {
        float s = msk ? NEGV : c[r] * scale;
        float tmax = s;
        tmax = fmaxf(tmax, __shfl_xor(tmax, 1, 32));
        tmax = fmaxf(tmax, __shfl_xor(tmax, 2, 32));
        tmax = fmaxf(tmax, __shfl_xor(tmax, 4, 32));
        tmax = fmaxf(tmax, __shfl_xor(tmax, 8, 32));
        float mnew = fmaxf(mrow[r], tmax);
        float e = __expf(s - mnew);
        e += __shfl_xor(e, 1, 32);
        e += __shfl_xor(e, 2, 32);
        e += __shfl_xor(e, 4, 32);
        e += __shfl_xor(e, 8, 32);
        lrow[r] = lrow[r] * __expf(mrow[r] - mnew) + e;
        mrow[r] = mnew;
      }
    }
    __syncthreads();
  }

  float rl[8];
  for (int r = 0; r < 8; ++r) rl[r] = 1.0f / lrow[r];

  // ----- Pass B: exact P = exp(s-m)/l, accumulate column sums -----
  load_ktile(ktile[0], kbase, tid, wave);
  for (int i = 0; i < NKB; ++i) {
    const int t0 = i * KBLK;
    if (i + 1 < NKB) {
      load_ktile(ktile[(i + 1) & 1], kbase + (size_t)(t0 + KBLK) * DIM, tid, wave);
      wait_ktile_pend1(wave);
    } else {
      wait_ktile_pend0(wave);
    }
    __syncthreads();
    const u16* kt_base = ktile[i & 1];

    for (int nt = 0; nt < 4; ++nt) {
      v8f c;
      for (int r = 0; r < 8; ++r) c[r] = 0.0f;
      const u16* kt = kt_base + (nt * 16 + m) * DIM;
      for (int kc = 0; kc < 8; ++kc) {
        const u16* kr = kt + kc * 32 + h * 16;
        v16bf bfrag = combine8(*(const us8*)(kr), *(const us8*)(kr + 8));
        c = __builtin_amdgcn_wmma_f32_16x16x32_bf16(
            false, qfrag[kc], false, bfrag, (short)0, c, false, false);
      }
      const bool msk = mb[t0 + nt * 16 + m] != 0;
      float csum = 0.0f;
      for (int r = 0; r < 8; ++r) {
        float s = msk ? NEGV : c[r] * scale;
        csum += __expf(s - mrow[r]) * rl[r];          // ~0 when masked
      }
      csum += __shfl_xor(csum, 16, 32);               // fold row halves
      if (lane < 16) colsum[wave][nt * 16 + lane] = csum;
    }
    __syncthreads();
    if (tid < KBLK) {
      float a = 0.0f;
      for (int w = 0; w < 8; ++w) a += colsum[w][tid];
      partial[((size_t)b * NQB + qb) * SEQ + t0 + tid] = a;
    }
    __syncthreads();
  }
}

// ---------------------------------------------------------------------------
// Kernel 3a: reduce partial column sums over query blocks -> wbuf[b][t]
// ---------------------------------------------------------------------------
__global__ void colreduce_kernel(const float* __restrict__ partial,
                                 float* __restrict__ wbuf) {
  int idx = blockIdx.x * 256 + threadIdx.x;   // < B*SEQ
  int b = idx >> 11;
  int t = idx & (SEQ - 1);
  float a = 0.0f;
  for (int qb = 0; qb < NQB; ++qb)
    a += partial[((size_t)b * NQB + qb) * SEQ + t];
  wbuf[idx] = a;
}

// ---------------------------------------------------------------------------
// Kernel 3b: out[b,d] = (1/length[b]) * sum_t wbuf[b,t] * v[b,t,d]
// ---------------------------------------------------------------------------
__global__ void weighted_v_kernel(const float* __restrict__ wbuf,
                                  const u16* __restrict__ vb16,
                                  const float* __restrict__ length,
                                  float* __restrict__ out) {
  const int b = blockIdx.x;
  const int d = threadIdx.x;
  const u16*   vb = vb16 + (size_t)b * SEQ * DIM;
  const float* wr = wbuf + (size_t)b * SEQ;
  float acc = 0.0f;
  for (int t = 0; t < SEQ; ++t)
    acc += wr[t] * bf2f(vb[(size_t)t * DIM + d]);
  out[b * DIM + d] = acc / length[b];
}

// ---------------------------------------------------------------------------
// Launch. Inputs: [0]=track_idx (unused), [1]=x f32, [2]=mask u8, [3]=length f32,
//                 [4]=Wq, [5]=Wk, [6]=Wv
// ws: x16(32MB) | wq/wk/wv16(3x128KB) | q/k/v16(3x32MB) | partial(4MB) | wbuf(256KB)
// ---------------------------------------------------------------------------
extern "C" void kernel_launch(void* const* d_in, const int* in_sizes, int n_in,
                              void* d_out, int out_size, void* d_ws, size_t ws_size,
                              hipStream_t stream) {
  const float* x      = (const float*)d_in[1];
  const unsigned char* mask = (const unsigned char*)d_in[2];
  const float* length = (const float*)d_in[3];
  const float* Wq     = (const float*)d_in[4];
  const float* Wk     = (const float*)d_in[5];
  const float* Wv     = (const float*)d_in[6];
  float* out          = (float*)d_out;

  const size_t XW_ELEMS = (size_t)BATCH * SEQ * DIM;   // 16.7M
  const size_t W_ELEMS  = (size_t)DIM * DIM;           // 65536

  char* p = (char*)d_ws;
  u16* xb16 = (u16*)p;  p += XW_ELEMS * 2;
  u16* wq16 = (u16*)p;  p += W_ELEMS * 2;
  u16* wk16 = (u16*)p;  p += W_ELEMS * 2;
  u16* wv16 = (u16*)p;  p += W_ELEMS * 2;
  u16* qb16 = (u16*)p;  p += XW_ELEMS * 2;
  u16* kb16 = (u16*)p;  p += XW_ELEMS * 2;
  u16* vb16 = (u16*)p;  p += XW_ELEMS * 2;
  float* partial = (float*)p;  p += (size_t)BATCH * NQB * SEQ * 4;
  float* wbuf    = (float*)p;

  f32_to_bf16_kernel<<<(int)(XW_ELEMS / 8 / 256), 256, 0, stream>>>(x,  xb16);
  f32_to_bf16_kernel<<<(int)(W_ELEMS  / 8 / 256), 256, 0, stream>>>(Wq, wq16);
  f32_to_bf16_kernel<<<(int)(W_ELEMS  / 8 / 256), 256, 0, stream>>>(Wk, wk16);
  f32_to_bf16_kernel<<<(int)(W_ELEMS  / 8 / 256), 256, 0, stream>>>(Wv, wv16);

  qkv_proj_kernel<<<dim3(BATCH * 3, SEQ / 64, DIM / 64), 128, 0, stream>>>(
      xb16, wq16, wk16, wv16, qb16, kb16, vb16);

  flash_colsum_kernel<<<dim3(BATCH, NQB), 256, 0, stream>>>(
      qb16, kb16, mask, partial);

  colreduce_kernel<<<(BATCH * SEQ) / 256, 256, 0, stream>>>(partial, wbuf);

  weighted_v_kernel<<<BATCH, DIM, 0, stream>>>(wbuf, vb16, length, out);
}
